// GraphAttentionModel_PyTorch_Geometric_35639638622885
// MI455X (gfx1250) — matrix operand order repaired and checked by reading, not compiled, not run on hardware
//
#include <hip/hip_runtime.h>
#include <hip/hip_bf16.h>
#include <math.h>

typedef __attribute__((ext_vector_type(16))) __bf16 v16bf;
typedef __attribute__((ext_vector_type(8)))  __bf16 v8bf;
typedef __attribute__((ext_vector_type(8)))  float  v8f;
typedef __attribute__((ext_vector_type(4)))  int    v4i;

#define NEG_SLOPE 0.2f

#if __has_builtin(__builtin_amdgcn_global_load_async_to_lds_b128)
#define HAVE_ASYNC_LDS 1
#else
#define HAVE_ASYNC_LDS 0
#endif

static __device__ __forceinline__ float lrelu(float x) { return x > 0.f ? x : NEG_SLOPE * x; }
// exp(x - nm) guarded so (-inf) - (-inf) never produces NaN
static __device__ __forceinline__ float expg(float x, float nm) {
    return (x == nm) ? 1.f : __expf(x - nm);
}

// 16-byte global -> LDS copy: async (ASYNCcnt-tracked) when available.
static __device__ __forceinline__ void copy16_g2l(const float* g, float* l) {
#if HAVE_ASYNC_LDS
    typedef __attribute__((address_space(1))) v4i* gp_t;
    typedef __attribute__((address_space(3))) v4i* lp_t;
    __builtin_amdgcn_global_load_async_to_lds_b128((gp_t)g, (lp_t)l, 0, 0);
#else
    *(float4*)l = *(const float4*)g;
#endif
}

template <int Nw>
static __device__ __forceinline__ void wait_async() {
#if HAVE_ASYNC_LDS
#if __has_builtin(__builtin_amdgcn_s_wait_asynccnt)
    __builtin_amdgcn_s_wait_asynccnt(Nw);
#else
    asm volatile("s_wait_asynccnt %0" ::"i"(Nw) : "memory");
#endif
#endif
}

// ---------------------------------------------------------------------------
// A-fragment: 16x32 bf16 from a row-major fp32 matrix.
// ISA layout: lane in [0,16): row = lane, elems 0..7 -> K = half*8 + e,
//             elems 8..15 -> K = 16 + half*8 + (e-8); half = lane>>4.
// ---------------------------------------------------------------------------
static __device__ __forceinline__ v16bf load_A_f32(const float* src, int row, int ld,
                                                   int kb, int half) {
    const float* p0 = src + (size_t)row * ld + kb + half * 8;
    const float* p1 = p0 + 16;
    v16bf a;
#pragma unroll
    for (int e = 0; e < 8; ++e) a[e] = (__bf16)p0[e];
#pragma unroll
    for (int e = 0; e < 8; ++e) a[8 + e] = (__bf16)p1[e];
    return a;
}

// B-fragment: 32x16 bf16 from row-major fp32 W[K x Ncols].
// ISA layout: lane n = lane&15 selects column, elems 0..7 -> K = half*8 + e,
// elems 8..15 -> K = 16 + half*8 + (e-8).
static __device__ __forceinline__ v16bf load_B_f32(const float* W, int ld, int kb,
                                                   int nb, int n, int half) {
    const float* p = W + (size_t)kb * ld + nb + n;
    v16bf b;
#pragma unroll
    for (int e = 0; e < 16; ++e) b[e] = (__bf16)p[(size_t)(((e >> 3) << 4) + half * 8 + (e & 7)) * ld];
    return b;
}

// ---------------------------------------------------------------------------
// GEMM: D = A(fp32, MxK row-major) x B(fp32, KxN row-major), bf16 WMMA,
// fp32 accumulate. Writes fp32 row-major result AND a bf16 transposed copy
// (outT is [N x M]) used later as contiguous V fragments.
// One wave computes one 16x16 tile.
// ---------------------------------------------------------------------------
__global__ void gemm_bf16_kernel(const float* __restrict__ A, const float* __restrict__ B,
                                 float* __restrict__ outRM, __bf16* __restrict__ outT,
                                 int M, int K, int Ncols) {
    const int lane = threadIdx.x & 31;
    const int wave = threadIdx.x >> 5;
    const int tiles_n = Ncols >> 4;
    const int tile = blockIdx.x * (blockDim.x >> 5) + wave;
    if (tile >= (M >> 4) * tiles_n) return;
    const int tm = tile / tiles_n, tn = tile % tiles_n;
    const int row0 = tm * 16, col0 = tn * 16;
    const int half = lane >> 4, li = lane & 15;

    v8f acc = {};
    for (int kb = 0; kb < K; kb += 32) {
        v16bf a = load_A_f32(A, row0 + li, K, kb, half);
        v16bf b = load_B_f32(B, Ncols, kb, col0, li, half);
        acc = __builtin_amdgcn_wmma_f32_16x16x32_bf16(false, a, false, b,
                                                      (short)0, acc, false, false);
    }
    // D layout: lane n = li (column), VGPR r -> row = half*8 + r
    const int mb = half * 8;
    v8bf tb;
#pragma unroll
    for (int r = 0; r < 8; ++r) {
        float v = acc[r];
        outRM[(size_t)(row0 + mb + r) * Ncols + col0 + li] = v;
        tb[r] = (__bf16)v;
    }
    // transposed bf16 store: 8 consecutive nodes in one row of outT -> b128
    *(v8bf*)(outT + (size_t)(col0 + li) * M + row0 + mb) = tb;
}

// ---------------------------------------------------------------------------
// a_src[n,h] = sum_c h[n,h,c]*att_src[h,c];  a_dst likewise. One thread/(n,h).
// ---------------------------------------------------------------------------
template <int H, int C>
__global__ void att_scores_kernel(const float* __restrict__ hfeat,
                                  const float* __restrict__ att_s,
                                  const float* __restrict__ att_d,
                                  float* __restrict__ asrc, float* __restrict__ adst, int N) {
    int idx = blockIdx.x * blockDim.x + threadIdx.x;
    if (idx >= N * H) return;
    int n = idx / H, hd = idx % H;
    const float* hp = hfeat + (size_t)n * (H * C) + hd * C;
    const float* sp = att_s + hd * C;
    const float* dp = att_d + hd * C;
    float ss = 0.f, dd = 0.f;
#pragma unroll 8
    for (int c = 0; c < C; ++c) {
        float v = hp[c];
        ss += v * sp[c];
        dd += v * dp[c];
    }
    asrc[idx] = ss;
    adst[idx] = dd;
}

// ---------------------------------------------------------------------------
// Softmax stats per destination row i (all heads): m = masked row max of
// lrelu(a_src[j]+a_dst[i]), s = sum exp(logit - m). One wave per row,
// lanes stride over j, butterfly-combined at the end.
// ---------------------------------------------------------------------------
template <int H>
__global__ void gat_stats_kernel(const float* __restrict__ adj,
                                 const float* __restrict__ asrc,
                                 const float* __restrict__ adst,
                                 float* __restrict__ mrow, float* __restrict__ srow, int N) {
    const int lane = threadIdx.x & 31;
    const int i = blockIdx.x * (blockDim.x >> 5) + (threadIdx.x >> 5);
    if (i >= N) return;

    float d[H], m[H], s[H];
#pragma unroll
    for (int h = 0; h < H; ++h) {
        d[h] = adst[(size_t)i * H + h];
        m[h] = -__builtin_inff();
        s[h] = 0.f;
    }
    for (int j = lane; j < N; j += 32) {
        float av = adj[(size_t)i * N + j];
        if (av != 0.f || j == i) {  // add_self_loops
#pragma unroll
            for (int h = 0; h < H; ++h) {
                float lg = lrelu(asrc[(size_t)j * H + h] + d[h]);
                float nm = fmaxf(m[h], lg);
                s[h] = s[h] * expg(m[h], nm) + expg(lg, nm);
                m[h] = nm;
            }
        }
    }
#pragma unroll
    for (int off = 16; off; off >>= 1) {
#pragma unroll
        for (int h = 0; h < H; ++h) {
            float mo = __shfl_xor(m[h], off, 32);
            float so = __shfl_xor(s[h], off, 32);
            float nm = fmaxf(m[h], mo);
            s[h] = s[h] * expg(m[h], nm) + so * expg(mo, nm);
            m[h] = nm;
        }
    }
    if (lane == 0) {
#pragma unroll
        for (int h = 0; h < H; ++h) {
            mrow[(size_t)i * H + h] = m[h];
            srow[(size_t)i * H + h] = s[h];
        }
    }
}

// ---------------------------------------------------------------------------
// Fused masked-softmax aggregation: out[i,h,:] = (1/s_i) * sum_j P[i,j]*V[j,:]
// with P = exp(lrelu(a_src[j]+a_dst[i]) - m_i) * mask. Block = 16 dst rows,
// one wave per head. adj tiles double-buffered in LDS via async
// global->LDS b128 copies (ASYNCcnt); P built straight into the 16x32 bf16
// A-fragment lane layout; V loaded from bf16-transposed features.
// APPLY_ELU=true writes elu(out+bias) (layer 1 -> x2), else out+bias (final).
// ---------------------------------------------------------------------------
template <int H, int C, bool APPLY_ELU>
__global__ void gat_aggregate_kernel(const float* __restrict__ adj,
                                     const float* __restrict__ asrc,
                                     const float* __restrict__ adst,
                                     const float* __restrict__ mrow,
                                     const float* __restrict__ srow,
                                     const __bf16* __restrict__ vT,
                                     const float* __restrict__ bias,
                                     float* __restrict__ outp, int N) {
    __shared__ float adjt[2][16 * 32];
    const int tid = threadIdx.x;
    const int lane = tid & 31;
    const int hd = tid >> 5;          // wave id == head
    const int half = lane >> 4, li = lane & 15;
    const int i0 = blockIdx.x * 16;
    const int irow = i0 + li;

    constexpr int BD = H * 32;              // blockDim.x
    constexpr int XFERS = 16 * 32 / 4;      // 128 b128 transfers per adj tile
    constexpr int MAXPT = (XFERS + BD - 1) / BD;  // per-thread issues per tile

    const float m_r = mrow[(size_t)irow * H + hd];
    const float s_r = srow[(size_t)irow * H + hd];
    const float inv_r = (s_r > 0.f) ? 1.f / s_r : 0.f;
    const float ad_r = adst[(size_t)irow * H + hd];

    v8f acc[C / 16] = {};

    auto stage = [&](int buf, int jc) {
        for (int t = tid; t < XFERS; t += BD) {
            int r = t >> 3, c4 = (t & 7) * 4;
            copy16_g2l(&adj[(size_t)(i0 + r) * N + jc + c4], &adjt[buf][r * 32 + c4]);
        }
    };

    stage(0, 0);
    for (int jc = 0, idx = 0; jc < N; jc += 32, ++idx) {
        const int cur = idx & 1;
        if (jc + 32 < N) stage(cur ^ 1, jc + 32);  // prefetch next tile (async)
        wait_async<MAXPT>();   // current tile's transfers done; next may be in flight
        __syncthreads();

        // Build P as A-fragment (16x32 bf16)
        const float* at = adjt[cur];
        v16bf a;
#pragma unroll
        for (int e = 0; e < 16; ++e) {
            int Kk = ((e >> 3) << 4) + half * 8 + (e & 7);
            int j = jc + Kk;
            float av = at[li * 32 + Kk];
            float p = 0.f;
            if (av != 0.f || j == irow) {
                float lg = lrelu(asrc[(size_t)j * H + hd] + ad_r);
                p = __expf(lg - m_r);   // m_r finite: self-loop guarantees a neighbor
            }
            a[e] = (__bf16)p;
        }
        // Multiply against C/16 V fragments (B: lane n=li column, K=((e>>3)<<4)+half*8+(e&7))
#pragma unroll
        for (int cc = 0; cc < C / 16; ++cc) {
            const __bf16* pv = vT + (size_t)(hd * C + cc * 16 + li) * N + jc + half * 8;
            v8bf lo = *(const v8bf*)pv;
            v8bf hi = *(const v8bf*)(pv + 16);
            v16bf b;
#pragma unroll
            for (int e = 0; e < 8; ++e) { b[e] = lo[e]; b[8 + e] = hi[e]; }
            acc[cc] = __builtin_amdgcn_wmma_f32_16x16x32_bf16(false, a, false, b,
                                                              (short)0, acc[cc], false, false);
        }
        __syncthreads();   // all reads of adjt[cur] done before it is restaged
    }

    // Epilogue: scale by 1/s (per D row, fetched via shuffle), bias, ELU/store.
    const int mb = half * 8;
    float invs[8];
#pragma unroll
    for (int r = 0; r < 8; ++r) invs[r] = __shfl(inv_r, mb + r, 32);
#pragma unroll
    for (int cc = 0; cc < C / 16; ++cc) {
#pragma unroll
        for (int r = 0; r < 8; ++r) {
            int col = hd * C + cc * 16 + li;
            float v = acc[cc][r] * invs[r] + bias[col];
            if (APPLY_ELU) v = (v > 0.f) ? v : (__expf(v) - 1.f);
            outp[(size_t)(i0 + mb + r) * (H * C) + col] = v;
        }
    }
}

// ---------------------------------------------------------------------------
extern "C" void kernel_launch(void* const* d_in, const int* in_sizes, int n_in,
                              void* d_out, int out_size, void* d_ws, size_t ws_size,
                              hipStream_t stream) {
    (void)in_sizes; (void)n_in; (void)out_size; (void)ws_size;
    const float* x   = (const float*)d_in[0];
    const float* adj = (const float*)d_in[1];
    const float* W1  = (const float*)d_in[2];
    const float* as1 = (const float*)d_in[3];
    const float* ad1 = (const float*)d_in[4];
    const float* b1  = (const float*)d_in[5];
    const float* W2  = (const float*)d_in[6];
    const float* as2 = (const float*)d_in[7];
    const float* ad2 = (const float*)d_in[8];
    const float* b2  = (const float*)d_in[9];
    float* out = (float*)d_out;

    const int N = 4096, F = 512, H = 8, C = 64, HC = 512, NC = 16;

    char* ws = (char*)d_ws;
    auto alloc = [&](size_t bytes) {
        char* p = ws;
        ws += (bytes + 255) & ~(size_t)255;
        return p;
    };
    float*  h1    = (float*)alloc((size_t)N * HC * sizeof(float));
    __bf16* h1T   = (__bf16*)alloc((size_t)HC * N * sizeof(__bf16));
    float*  x2    = (float*)alloc((size_t)N * HC * sizeof(float));
    float*  p2    = (float*)alloc((size_t)N * NC * sizeof(float));
    __bf16* p2T   = (__bf16*)alloc((size_t)NC * N * sizeof(__bf16));
    float*  asrc1 = (float*)alloc((size_t)N * H * sizeof(float));
    float*  adst1 = (float*)alloc((size_t)N * H * sizeof(float));
    float*  m1    = (float*)alloc((size_t)N * H * sizeof(float));
    float*  s1    = (float*)alloc((size_t)N * H * sizeof(float));
    float*  asrc2 = (float*)alloc((size_t)N * sizeof(float));
    float*  adst2 = (float*)alloc((size_t)N * sizeof(float));
    float*  m2    = (float*)alloc((size_t)N * sizeof(float));
    float*  s2    = (float*)alloc((size_t)N * sizeof(float));

    // ---- Layer 1 ----
    {   // h1 = x @ W1  (8192 tiles, 8 waves/block)
        int tiles = (N / 16) * (HC / 16);
        gemm_bf16_kernel<<<tiles / 8, 256, 0, stream>>>(x, W1, h1, h1T, N, F, HC);
    }
    att_scores_kernel<H, C><<<(N * H + 255) / 256, 256, 0, stream>>>(h1, as1, ad1, asrc1, adst1, N);
    gat_stats_kernel<H><<<N / 8, 256, 0, stream>>>(adj, asrc1, adst1, m1, s1, N);
    gat_aggregate_kernel<H, C, true><<<N / 16, H * 32, 0, stream>>>(adj, asrc1, adst1, m1, s1,
                                                                    h1T, b1, x2, N);
    // ---- Layer 2 ----
    {   // p2 = x2 @ W2  (256 tiles)
        int tiles = (N / 16) * (NC / 16);
        gemm_bf16_kernel<<<tiles / 8, 256, 0, stream>>>(x2, W2, p2, p2T, N, HC, NC);
    }
    att_scores_kernel<1, NC><<<(N + 255) / 256, 256, 0, stream>>>(p2, as2, ad2, asrc2, adst2, N);
    gat_stats_kernel<1><<<N / 8, 256, 0, stream>>>(adj, asrc2, adst2, m2, s2, N);
    gat_aggregate_kernel<1, NC, false><<<N / 16, 32, 0, stream>>>(adj, asrc2, adst2, m2, s2,
                                                                  p2T, b2, out, N);
}
